// FewShotGNN_4784593568171
// MI455X (gfx1250) — matrix operand order repaired and checked by reading, not compiled, not run on hardware
//
#include <hip/hip_runtime.h>
#include <hip/hip_bf16.h>

typedef __attribute__((ext_vector_type(16))) __bf16 v16bf;
typedef __attribute__((ext_vector_type(8)))  float  v8f;
typedef unsigned int u32x4 __attribute__((ext_vector_type(4)));
typedef int          i32x4 __attribute__((ext_vector_type(4)));
typedef int          i32x8 __attribute__((ext_vector_type(8)));

#define N_NODES 50000
#define F_IN    512
#define F_HID   256
#define F_OUT   128

#define KC       64                   // K-chunk staged per TDM transfer
#define BCOLS    128                  // B columns staged per block
#define BROWPAD  144                  // bytes per staged B column (64*2 + 16 pad)
#define BBUF_SZ  (BCOLS * BROWPAD)    // 18432 B per buffer
#define SMEM_SZ  (2 * BBUF_SZ)        // double buffered

// ---------------- workspace layout (bytes, 256-aligned) ----------------
static const size_t OFF_DEG   = 0;
static const size_t OFF_NORM  = 200192;                 // 50000*4 -> 200192
static const size_t OFF_W1T   = OFF_NORM + 3200000;     // 3400192
static const size_t OFF_W2T   = OFF_W1T + 262144;       // 3662336
static const size_t OFF_XB    = OFF_W2T + 65536;        // 3727872
static const size_t OFF_H1    = OFF_XB + 51200000;      // 54927872
static const size_t OFF_AGG1  = OFF_H1 + 51200000;      // 106127872
static const size_t OFF_RELUB = OFF_XB;                  // alias (x_bf16 dead after GEMM1)
static const size_t OFF_H2    = OFF_XB + 25600000;       // alias upper half
static const size_t OFF_AGG2  = OFF_H1;                  // alias (H1 dead after finalize1)

// ---------------- small utility kernels ----------------
__global__ void k_f32_to_bf16(const float* __restrict__ in, __bf16* __restrict__ out, int n) {
    int i = blockIdx.x * blockDim.x + threadIdx.x;
    if (i < n) out[i] = (__bf16)in[i];
}

// Wt[n*K + k] = W[k*Nc + n]  (K-major so TDM rows and B-fragment K runs are contiguous)
__global__ void k_transpose_bf16(const float* __restrict__ W, __bf16* __restrict__ Wt, int K, int Nc) {
    int i = blockIdx.x * blockDim.x + threadIdx.x;
    if (i >= K * Nc) return;
    int k = i / Nc, n = i - k * Nc;
    Wt[(size_t)n * K + k] = (__bf16)W[i];
}

__global__ void k_degree(const int* __restrict__ col, const float* __restrict__ ew,
                         float* __restrict__ deg, int E) {
    int e = blockIdx.x * blockDim.x + threadIdx.x;
    if (e < E) atomicAdd(&deg[col[e]], ew[e]);
}

__global__ void k_dinv(float* __restrict__ d, int n) {
    int i = blockIdx.x * blockDim.x + threadIdx.x;
    if (i < n) d[i] = rsqrtf(d[i] + 1.0f);   // +1.0 = self-loop weight
}

__global__ void k_norm(const int* __restrict__ row, const int* __restrict__ col,
                       const float* __restrict__ ew, const float* __restrict__ dinv,
                       float* __restrict__ nrm, int E) {
    int e = blockIdx.x * blockDim.x + threadIdx.x;
    if (e < E) nrm[e] = dinv[row[e]] * ew[e] * dinv[col[e]];
}

// ---------------- TDM descriptor: DMA a [BCOLS x KC] bf16 tile of Bt into LDS ----------------
// Tile rows = Bt rows (output columns), row length KC elements, row stride K elements.
// LDS padding: 4 DWORDs (16B) after every 32 DWORDs (128B = one KC row) -> 144 B/col.
__device__ __forceinline__ void tdm_load_b(unsigned int lds_off, const __bf16* gptr, int K, int Nc) {
    unsigned long long ga = (unsigned long long)(uintptr_t)gptr;
    union { u32x4 v; unsigned int u[4]; } g0;
    g0.u[0] = 1u;                                            // count=1, user mode, no gather
    g0.u[1] = lds_off;                                       // lds_addr [63:32]
    g0.u[2] = (unsigned int)ga;                              // global_addr [95:64]
    g0.u[3] = (unsigned int)((ga >> 32) & 0x1FFFFFFu)        // global_addr [120:96]
            | (2u << 30);                                    // type=2 ("image")
    union { i32x8 v; unsigned int u[8]; } g1;
    g1.u[0] = (1u << 16)          // data_size = 2 bytes
            | (1u << 20)          // pad_enable
            | (4u << 22)          // pad_interval: 32 DWORDs
            | (3u << 25);         // pad_amount: 4 DWORDs (16 B)
    g1.u[1] = ((unsigned int)K & 0xFFFFu) << 16;             // tensor_dim0 [79:48] lo
    g1.u[2] = ((unsigned int)K >> 16)                        // tensor_dim0 hi
            | (((unsigned int)Nc & 0xFFFFu) << 16);          // tensor_dim1 [111:80] lo
    g1.u[3] = ((unsigned int)Nc >> 16)                       // tensor_dim1 hi
            | ((unsigned int)KC << 16);                      // tile_dim0 [127:112]
    g1.u[4] = (unsigned int)BCOLS;                           // tile_dim1 [143:128]
    g1.u[5] = (unsigned int)K;                               // tensor_dim0_stride lo32
    g1.u[6] = 0u;                                            // stride hi16 | dim1_stride lo
    g1.u[7] = 0u;
    i32x4 z4 = {0, 0, 0, 0};
#if __clang_major__ >= 23
    i32x8 z8 = {0, 0, 0, 0, 0, 0, 0, 0};
    __builtin_amdgcn_tensor_load_to_lds(g0.v, g1.v, z4, z4, z8, 0);
#else
    __builtin_amdgcn_tensor_load_to_lds(g0.v, g1.v, z4, z4, 0);
#endif
}

// ---------------- WMMA GEMM with TDM-staged, double-buffered B tiles ----------------
// Block: 8 waves x 16 rows = 128 rows, BCOLS=128 cols. Wave: 16x128 strip (8 accumulators).
// Inner K-step: issue ALL 8 B-fragment LDS loads first, then the 8 WMMAs, so the compiler
// can emit one deep ds_load clause and overlap loads with WMMA via non-zero dscnt waits.
__global__ __launch_bounds__(256) void k_gemm_bf16_tdm(const __bf16* __restrict__ A,
                                                       const __bf16* __restrict__ Bt,
                                                       float* __restrict__ C,
                                                       int M, int K, int Nc) {
    extern __shared__ char smem[];                 // SMEM_SZ bytes, 2 x B buffer
    const int lane = threadIdx.x & 31;
    const int wave = threadIdx.x >> 5;
    const int rt   = blockIdx.y * 8 + wave;        // 16-row tile index (wave-uniform)
    const int row0 = rt * 16;
    const bool active = (row0 < M);                // uniform per wave; inactive waves still barrier
    const int colb = blockIdx.x * BCOLS;

    // A fragment (16-bit A 16x32): lane<16 -> K 0..7 & 16..23; lane>=16 -> K 8..15 & 24..31
    const int mrow = active ? (row0 + (lane & 15)) : (lane & 15);
    const int koff = (lane < 16) ? 0 : 8;
    const __bf16* __restrict__ arow = A + (size_t)mrow * K;

    const unsigned int lds_base = (unsigned int)(uintptr_t)smem;  // LDS byte offset (low 32 bits)

    // prologue: DMA chunk 0 into buffer 0
    if (wave == 0) {
        tdm_load_b(lds_base, Bt + (size_t)colb * K, K, Nc);
        __builtin_amdgcn_s_wait_tensorcnt(0);
    }
    __syncthreads();

    v8f acc[8];
#pragma unroll
    for (int j = 0; j < 8; ++j)
#pragma unroll
        for (int i = 0; i < 8; ++i) acc[j][i] = 0.0f;

    const int nchunk = K / KC;
    for (int c = 0; c < nchunk; ++c) {
        // prefetch next chunk into the other buffer while computing on this one
        if (wave == 0 && (c + 1) < nchunk) {
            tdm_load_b(lds_base + (unsigned int)(((c + 1) & 1) * BBUF_SZ),
                       Bt + (size_t)colb * K + (size_t)(c + 1) * KC, K, Nc);
        }
        const char* __restrict__ bbuf = smem + (c & 1) * BBUF_SZ;
        const int kc = c * KC;
#pragma unroll
        for (int ks = 0; ks < KC; ks += 32) {
            union { v16bf v; uint4 q[2]; } a;
            a.q[0] = *(const uint4*)(arow + kc + ks + koff);
            a.q[1] = *(const uint4*)(arow + kc + ks + 16 + koff);
            const int kb = ks + ((lane < 16) ? 0 : 8);     // K base within chunk for B frag
            const char* bcol = bbuf + (size_t)(lane & 15) * BROWPAD + (size_t)kb * 2;
            union { v16bf v; uint4 q[2]; } b[8];
#pragma unroll
            for (int j = 0; j < 8; ++j) {                  // issue all LDS loads first
                b[j].q[0] = *(const uint4*)(bcol + j * (16 * BROWPAD));
                b[j].q[1] = *(const uint4*)(bcol + j * (16 * BROWPAD) + 32);
            }
#pragma unroll
            for (int j = 0; j < 8; ++j) {                  // then all WMMAs
                acc[j] = __builtin_amdgcn_wmma_f32_16x16x32_bf16(
                    false, a.v, false, b[j].v, (short)0, acc[j], false, false);
            }
        }
        if (wave == 0) __builtin_amdgcn_s_wait_tensorcnt(0);  // prefetch landed
        __syncthreads();                                       // all waves done with bbuf
    }

    if (active) {
        // C layout: VGPR r -> (lane<16: M=row0+r, N=lane) (lane>=16: M=row0+8+r, N=lane-16)
        const int srow = row0 + ((lane < 16) ? 0 : 8);
        const int scol = lane & 15;
#pragma unroll
        for (int j = 0; j < 8; ++j) {
            float* __restrict__ cc = C + (size_t)srow * Nc + colb + j * 16 + scol;
#pragma unroll
            for (int r = 0; r < 8; ++r) cc[(size_t)r * Nc] = acc[j][r];
        }
    }
}

// ---------------- edge aggregation: out[col[e],:] += norm[e] * H[row[e],:] ----------------
__global__ void k_aggregate(const float* __restrict__ H, const int* __restrict__ row,
                            const int* __restrict__ col, const float* __restrict__ nrm,
                            float* __restrict__ out, int E, int F) {
    int wid  = (blockIdx.x * blockDim.x + threadIdx.x) >> 5;
    int lane = threadIdx.x & 31;
    if (wid >= E) return;
    int   r  = row[wid];
    int   c  = col[wid];
    float nv = nrm[wid];
    const float* __restrict__ hr = H + (size_t)r * F;
    float* __restrict__       oc = out + (size_t)c * F;
    for (int f = lane * 4; f < F; f += 128) {
        float4 v = *(const float4*)(hr + f);
        atomicAdd(oc + f + 0, v.x * nv);
        atomicAdd(oc + f + 1, v.y * nv);
        atomicAdd(oc + f + 2, v.z * nv);
        atomicAdd(oc + f + 3, v.w * nv);
    }
}

// finalize layer1: relu(agg + dinv^2*H (self-loop) + b) -> bf16 for GEMM2
__global__ void k_finalize1(const float* __restrict__ agg, const float* __restrict__ H,
                            const float* __restrict__ dinv, const float* __restrict__ b,
                            __bf16* __restrict__ out, int total, int F) {
    int i = blockIdx.x * blockDim.x + threadIdx.x;
    if (i >= total) return;
    int   n = i / F, f = i - n * F;
    float d = dinv[n];
    float v = agg[i] + d * d * H[i] + b[f];
    out[i] = (__bf16)fmaxf(v, 0.0f);
}

// finalize layer2: v = agg + dinv^2*H + b, then row-wise L2 normalize. One wave per row.
__global__ void k_finalize2(const float* __restrict__ agg, const float* __restrict__ H,
                            const float* __restrict__ dinv, const float* __restrict__ b,
                            float* __restrict__ out, int Nrows) {
    int gw   = (blockIdx.x * blockDim.x + threadIdx.x) >> 5;
    int lane = threadIdx.x & 31;
    if (gw >= Nrows) return;
    const size_t base = (size_t)gw * F_OUT;
    float d2 = dinv[gw] * dinv[gw];
    float vals[F_OUT / 32];
    float ss = 0.0f;
#pragma unroll
    for (int j = 0; j < F_OUT / 32; ++j) {
        int   f = lane + j * 32;
        float v = agg[base + f] + d2 * H[base + f] + b[f];
        vals[j] = v;
        ss += v * v;
    }
#pragma unroll
    for (int m = 16; m >= 1; m >>= 1) ss += __shfl_xor(ss, m, 32);
    float sc = 1.0f / fmaxf(sqrtf(ss), 1e-12f);
#pragma unroll
    for (int j = 0; j < F_OUT / 32; ++j) out[base + lane + j * 32] = vals[j] * sc;
}

// ---------------- host-side launch ----------------
extern "C" void kernel_launch(void* const* d_in, const int* in_sizes, int n_in,
                              void* d_out, int out_size, void* d_ws, size_t ws_size,
                              hipStream_t stream) {
    const float* x   = (const float*)d_in[0];
    const int*   ei  = (const int*)d_in[1];
    const float* ew  = (const float*)d_in[2];
    const float* W1  = (const float*)d_in[3];
    const float* b1  = (const float*)d_in[4];
    const float* W2  = (const float*)d_in[5];
    const float* b2  = (const float*)d_in[6];
    const int E = in_sizes[2];
    const int* row = ei;        // edge_index[0]
    const int* col = ei + E;    // edge_index[1]

    char* ws = (char*)d_ws;
    float*  deg   = (float*)(ws + OFF_DEG);     // becomes dinv in place
    float*  nrm   = (float*)(ws + OFF_NORM);
    __bf16* w1t   = (__bf16*)(ws + OFF_W1T);
    __bf16* w2t   = (__bf16*)(ws + OFF_W2T);
    __bf16* xb    = (__bf16*)(ws + OFF_XB);
    float*  h1    = (float*)(ws + OFF_H1);
    float*  agg1  = (float*)(ws + OFF_AGG1);
    __bf16* relub = (__bf16*)(ws + OFF_RELUB);
    float*  h2    = (float*)(ws + OFF_H2);
    float*  agg2  = (float*)(ws + OFF_AGG2);
    float*  out   = (float*)d_out;

    // --- degree / dinv / edge norms ---
    hipMemsetAsync(deg, 0, (size_t)N_NODES * 4, stream);
    k_degree<<<(E + 255) / 256, 256, 0, stream>>>(col, ew, deg, E);
    k_dinv<<<(N_NODES + 255) / 256, 256, 0, stream>>>(deg, N_NODES);
    k_norm<<<(E + 255) / 256, 256, 0, stream>>>(row, col, ew, deg, nrm, E);

    // --- bf16 conversions ---
    {
        int n = N_NODES * F_IN;
        k_f32_to_bf16<<<(n + 255) / 256, 256, 0, stream>>>(x, xb, n);
    }
    k_transpose_bf16<<<(F_IN * F_HID + 255) / 256, 256, 0, stream>>>(W1, w1t, F_IN, F_HID);
    k_transpose_bf16<<<(F_HID * F_OUT + 255) / 256, 256, 0, stream>>>(W2, w2t, F_HID, F_OUT);

    const int ytiles = (N_NODES / 16 + 7) / 8;   // 391 blocks of 128 rows

    // --- layer 1: GEMM (WMMA bf16, TDM double-buffered B) ---
    {
        dim3 grid(F_HID / BCOLS, ytiles);        // (2, 391)
        k_gemm_bf16_tdm<<<grid, 256, SMEM_SZ, stream>>>(xb, w1t, h1, N_NODES, F_IN, F_HID);
    }
    // --- layer 1: aggregation + finalize (bias, self-loop, relu -> bf16) ---
    hipMemsetAsync(agg1, 0, (size_t)N_NODES * F_HID * 4, stream);
    k_aggregate<<<(E * 32 + 255) / 256, 256, 0, stream>>>(h1, row, col, nrm, agg1, E, F_HID);
    {
        int n = N_NODES * F_HID;
        k_finalize1<<<(n + 255) / 256, 256, 0, stream>>>(agg1, h1, deg, b1, relub, n, F_HID);
    }

    // --- layer 2: GEMM (WMMA bf16, TDM double-buffered B) ---
    {
        dim3 grid(F_OUT / BCOLS, ytiles);        // (1, 391)
        k_gemm_bf16_tdm<<<grid, 256, SMEM_SZ, stream>>>(relub, w2t, h2, N_NODES, F_HID, F_OUT);
    }
    // --- layer 2: aggregation + finalize (bias, self-loop, L2 normalize) ---
    hipMemsetAsync(agg2, 0, (size_t)N_NODES * F_OUT * 4, stream);
    k_aggregate<<<(E * 32 + 255) / 256, 256, 0, stream>>>(h2, row, col, nrm, agg2, E, F_OUT);
    k_finalize2<<<(N_NODES * 32 + 255) / 256, 256, 0, stream>>>(agg2, h2, deg, b2, out, N_NODES);

    (void)n_in; (void)out_size; (void)ws_size;
}
